// HeirarchicalMoE_45011257262409
// MI455X (gfx1250) — matrix operand order repaired and checked by reading, not compiled, not run on hardware
//
#include <hip/hip_runtime.h>
#include <hip/hip_bf16.h>

// ---------------- problem constants (from setup_inputs) ----------------
constexpr int B_   = 8;
constexpr int N_   = 2048;
constexpr int D_   = 1024;
constexpr int EO_  = 4;
constexpr int EI_  = 4;
constexpr int H_   = 4096;
constexpr int CAPO = 640;                 // max(4, min(2048, int(2048*1.25/4)))
constexpr int CAPI = 200;                 // max(4, min(640,  int(640*1.25/4)))
constexpr int TOK  = B_ * N_;             // 16384 tokens
constexpr int SLO  = EO_ * B_ * CAPO;     // 20480 outer slots
constexpr int SLI  = EO_ * EI_ * B_ * CAPI; // 25600 inner slots
constexpr int MR   = B_ * CAPI;           // 1600 GEMM rows per (eo,ei)

// ---------------- CDNA5 WMMA types ----------------
typedef __attribute__((ext_vector_type(16))) __bf16 v16bf;
typedef __attribute__((ext_vector_type(8)))  float  v8f;
typedef __attribute__((ext_vector_type(4)))  unsigned int uint4v;
typedef __attribute__((ext_vector_type(8)))  unsigned int uint8v;
union Frag16 { unsigned int u[8]; v16bf v; };

__device__ __forceinline__ unsigned short f2bf(float f) {
  unsigned int u = __float_as_uint(f);
  unsigned int r = u + 0x7fffu + ((u >> 16) & 1u);   // round-to-nearest-even
  return (unsigned short)(r >> 16);
}
__device__ __forceinline__ unsigned int pack2bf(float a, float b) {
  return (unsigned int)f2bf(a) | ((unsigned int)f2bf(b) << 16);
}

// ---------------- CDNA5 async / TDM helpers ----------------
// Per-lane async DMA: 16B global -> LDS, tracked by ASYNCcnt.
__device__ __forceinline__ void async_ld_b128(unsigned lds_off, const void* gptr) {
  asm volatile("global_load_async_to_lds_b128 %0, %1, off"
               :: "v"(lds_off), "v"(gptr) : "memory");
}
__device__ __forceinline__ void wait_async0() {
  asm volatile("s_wait_asynccnt 0" ::: "memory");
}

// Tensor Data Mover: 2D tile (tile_d1 rows x tile_d0 dwords, row stride `stride0`
// dwords in global) -> LDS at lds_off, with optional LDS row padding
// (pad_interval code 6 = 128 dwords, pad_amount code 3 = 4 dwords -> 132-dword rows).
__device__ __forceinline__ void tdm_load_2d(unsigned lds_off, const void* gaddr,
                                            unsigned tile_d0, unsigned tile_d1,
                                            unsigned stride0) {
  unsigned long long ga = (unsigned long long)gaddr;
  uint4v g0;
  g0[0] = 1u;                                          // count=1 (valid user D#)
  g0[1] = lds_off;                                     // LDS byte address
  g0[2] = (unsigned)(ga & 0xffffffffu);                // global_addr[31:0]
  g0[3] = (unsigned)((ga >> 32) & 0x01ffffffu) | (2u << 30); // addr[56:32], type=2
  uint8v g1;
  g1[0] = (2u << 16) | (1u << 20) | (6u << 22) | (3u << 25); // 4B elems, pad 4dw/128dw
  g1[1] = (tile_d0 & 0xffffu) << 16;                   // tensor_dim0[15:0]  (== tile)
  g1[2] = (tile_d0 >> 16) | ((tile_d1 & 0xffffu) << 16); // dim0[31:16] | dim1[15:0]
  g1[3] = (tile_d1 >> 16) | (tile_d0 << 16);           // dim1[31:16] | tile_dim0
  g1[4] = (tile_d1 & 0xffffu);                         // tile_dim1 (tile_dim2 = 0)
  g1[5] = stride0;                                     // tensor_dim0_stride[31:0]
  g1[6] = 0u;                                          // stride0[47:32] | stride1 lo
  g1[7] = 0u;
  asm volatile("tensor_load_to_lds %0, %1" :: "s"(g0), "s"(g1) : "memory");
}

// ---------------- init: sparse maps to -1, slot gates to 0 ----------------
__global__ void k_init(int* __restrict__ slot_src_o, float* __restrict__ slot_gate_o,
                       int* __restrict__ ir1, int* __restrict__ ir2,
                       int* __restrict__ slot_src_i) {
  int i = blockIdx.x * 256 + threadIdx.x;
  if (i < SLO) { slot_src_o[i] = -1; slot_gate_o[i] = 0.f; ir1[i] = -1; ir2[i] = -1; }
  if (i < SLI) { slot_src_i[i] = -1; }
}

// ---------------- outer gating: logits, softmax, top-2 ----------------
__global__ void __launch_bounds__(128) k_outer_gate(
    const float* __restrict__ x, const float* __restrict__ wg,
    int* __restrict__ e1o, int* __restrict__ e2o,
    float* __restrict__ g1o, float* __restrict__ g2o,
    float4* __restrict__ praw) {
  int t = blockIdx.x * 128 + threadIdx.x;
  if (t >= TOK) return;
  const float* xr = x + (size_t)t * D_;
  float l0 = 0.f, l1 = 0.f, l2 = 0.f, l3 = 0.f;
  for (int d = 0; d < D_; d += 4) {
    float4 xv = *(const float4*)(xr + d);
    float4 w0 = *(const float4*)(wg + (d + 0) * EO_);
    float4 w1 = *(const float4*)(wg + (d + 1) * EO_);
    float4 w2 = *(const float4*)(wg + (d + 2) * EO_);
    float4 w3 = *(const float4*)(wg + (d + 3) * EO_);
    l0 += xv.x * w0.x + xv.y * w1.x + xv.z * w2.x + xv.w * w3.x;
    l1 += xv.x * w0.y + xv.y * w1.y + xv.z * w2.y + xv.w * w3.y;
    l2 += xv.x * w0.z + xv.y * w1.z + xv.z * w2.z + xv.w * w3.z;
    l3 += xv.x * w0.w + xv.y * w1.w + xv.z * w2.w + xv.w * w3.w;
  }
  float l[4] = {l0, l1, l2, l3};
  float mx = fmaxf(fmaxf(l[0], l[1]), fmaxf(l[2], l[3]));
  float p[4], s = 0.f;
  for (int i = 0; i < 4; ++i) { p[i] = __expf(l[i] - mx); s += p[i]; }
  for (int i = 0; i < 4; ++i) p[i] /= s;
  int e1 = 0; float g1 = p[0];
  for (int i = 1; i < 4; ++i) if (p[i] > g1) { g1 = p[i]; e1 = i; }
  float q[4] = {p[0], p[1], p[2], p[3]};
  q[e1] = 0.f;
  int e2 = 0; float g2 = q[0];
  for (int i = 1; i < 4; ++i) if (q[i] > g2) { g2 = q[i]; e2 = i; }
  float dn = g1 + g2 + 1e-9f;
  e1o[t] = e1; e2o[t] = e2; g1o[t] = g1 / dn; g2o[t] = g2 / dn;
  praw[t] = make_float4(p[0], p[1], p[2], p[3]);
}

// ---------------- outer position scan (exact exclusive-cumsum semantics) ----------------
__global__ void k_outer_scan(
    const int* __restrict__ e1, const int* __restrict__ e2,
    const float* __restrict__ g1, const float* __restrict__ g2,
    const float4* __restrict__ praw,
    int* __restrict__ p1, int* __restrict__ p2,
    int* __restrict__ slot_src, float* __restrict__ slot_gate,
    float* __restrict__ proxy_o, float* __restrict__ cnt_o) {
  int b = blockIdx.x;
  int e = threadIdx.x;
  if (e >= EO_) return;
  float psum = 0.f, csum = 0.f;
  int cnt = 0;
  for (int n = 0; n < N_; ++n) {
    int t = b * N_ + n;
    float4 pr = praw[t];
    psum += (e == 0 ? pr.x : e == 1 ? pr.y : e == 2 ? pr.z : pr.w);
    if (e1[t] == e) {
      csum += 1.f;
      int p = cnt++;
      if (p < CAPO) {
        p1[t] = p;
        int sIdx = (e * B_ + b) * CAPO + p;
        slot_src[sIdx] = n; slot_gate[sIdx] = g1[t];
      } else p1[t] = -1;
    }
  }
  int base = cnt < CAPO ? cnt : CAPO;  // truncated mask_1_count
  int c2 = 0;
  for (int n = 0; n < N_; ++n) {
    int t = b * N_ + n;
    if (e2[t] == e) {
      int p = base + c2++;
      if (p < CAPO) {
        p2[t] = p;
        if (g2[t] > 0.f) {
          int sIdx = (e * B_ + b) * CAPO + p;
          slot_src[sIdx] = n; slot_gate[sIdx] = g2[t];
        }
      } else p2[t] = -1;
    }
  }
  proxy_o[b * EO_ + e] = psum;
  cnt_o[b * EO_ + e]   = csum;
}

// ---------------- gather dispatched tokens (fp32, zero empty slots) ----------------
__global__ void __launch_bounds__(256) k_gather_outer(
    const float* __restrict__ x, const int* __restrict__ slot_src,
    float* __restrict__ exp_in_o) {
  int row = blockIdx.x;                          // [EO][B][CAPO]
  int b   = (row / CAPO) % B_;
  int src = slot_src[row];
  int d   = threadIdx.x * 4;
  float4 v = make_float4(0.f, 0.f, 0.f, 0.f);
  if (src >= 0) v = *(const float4*)(x + ((size_t)b * N_ + src) * D_ + d);
  *(float4*)(exp_in_o + (size_t)row * D_ + d) = v;
}

// ---------------- inner gating with importance masking ----------------
__global__ void __launch_bounds__(128) k_inner_gate(
    const float* __restrict__ xin, const float* __restrict__ wgi,
    const float* __restrict__ slot_gate,
    int* __restrict__ ie1, int* __restrict__ ie2,
    float* __restrict__ ig1, float* __restrict__ ig2,
    float4* __restrict__ praw_i) {
  int row = blockIdx.x * 128 + threadIdx.x;
  if (row >= SLO) return;
  int e = row / (B_ * CAPO);
  const float* xr  = xin + (size_t)row * D_;
  const float* wge = wgi + (size_t)e * D_ * EI_;
  float l0 = 0.f, l1 = 0.f, l2 = 0.f, l3 = 0.f;
  for (int d = 0; d < D_; d += 4) {
    float4 xv = *(const float4*)(xr + d);
    float4 w0 = *(const float4*)(wge + (d + 0) * EI_);
    float4 w1 = *(const float4*)(wge + (d + 1) * EI_);
    float4 w2 = *(const float4*)(wge + (d + 2) * EI_);
    float4 w3 = *(const float4*)(wge + (d + 3) * EI_);
    l0 += xv.x * w0.x + xv.y * w1.x + xv.z * w2.x + xv.w * w3.x;
    l1 += xv.x * w0.y + xv.y * w1.y + xv.z * w2.y + xv.w * w3.y;
    l2 += xv.x * w0.z + xv.y * w1.z + xv.z * w2.z + xv.w * w3.z;
    l3 += xv.x * w0.w + xv.y * w1.w + xv.z * w2.w + xv.w * w3.w;
  }
  float l[4] = {l0, l1, l2, l3};
  float mx = fmaxf(fmaxf(l[0], l[1]), fmaxf(l[2], l[3]));
  float p[4], s = 0.f;
  for (int i = 0; i < 4; ++i) { p[i] = __expf(l[i] - mx); s += p[i]; }
  for (int i = 0; i < 4; ++i) p[i] /= s;

  float sg  = slot_gate[row];
  float imp = (sg > 0.5f) ? 1.0f : (sg > 0.f ? 0.5f : 0.f);
  bool  m1  = (imp == 1.0f);
  bool  m2  = (imp > 0.f);

  int e1 = 0; float g1r = p[0];
  for (int i = 1; i < 4; ++i) if (p[i] > g1r) { g1r = p[i]; e1 = i; }
  float q[4] = {p[0], p[1], p[2], p[3]};
  if (m1) q[e1] = 0.f;                 // mask_1 already includes eq_one
  int e2 = 0; float g2r = q[0];
  for (int i = 1; i < 4; ++i) if (q[i] > g2r) { g2r = q[i]; e2 = i; }

  float G1 = m1 ? g1r : 0.f;
  float dn = G1 + g2r + 1e-9f;
  G1 /= dn;
  float G2 = g2r / dn;
  ie1[row] = m1 ? e1 : -1;
  ie2[row] = m2 ? e2 : -1;
  ig1[row] = G1; ig2[row] = G2;
  praw_i[row] = m1 ? make_float4(p[0], p[1], p[2], p[3])
                   : make_float4(0.f, 0.f, 0.f, 0.f);
}

// ---------------- inner position scan ----------------
__global__ void k_inner_scan(
    const int* __restrict__ ie1, const int* __restrict__ ie2,
    const float* __restrict__ ig2, const float4* __restrict__ praw_i,
    int* __restrict__ ir1, int* __restrict__ ir2, int* __restrict__ slot_src_i,
    float* __restrict__ proxy_i, float* __restrict__ cnt_i) {
  int grp = blockIdx.x;                // e*B_ + b
  int e = grp / B_, b = grp % B_;
  int f = threadIdx.x;
  if (f >= EI_) return;
  float psum = 0.f, csum = 0.f;
  int cnt = 0;
  for (int c = 0; c < CAPO; ++c) {
    int row = grp * CAPO + c;
    float4 pr = praw_i[row];
    psum += (f == 0 ? pr.x : f == 1 ? pr.y : f == 2 ? pr.z : pr.w);
    if (ie1[row] == f) {
      csum += 1.f;
      int p = cnt++;
      if (p < CAPI) {
        int r = ((e * EI_ + f) * B_ + b) * CAPI + p;
        ir1[row] = r; slot_src_i[r] = row;
      } else ir1[row] = -1;
    }
  }
  int base = cnt < CAPI ? cnt : CAPI;
  int c2 = 0;
  for (int c = 0; c < CAPO; ++c) {
    int row = grp * CAPO + c;
    if (ie2[row] == f) {
      int p = base + c2++;
      if (p < CAPI && ig2[row] > 0.f) {
        int r = ((e * EI_ + f) * B_ + b) * CAPI + p;
        ir2[row] = r; slot_src_i[r] = row;
      }
    }
  }
  proxy_i[grp * EI_ + f] = psum;
  cnt_i[grp * EI_ + f]   = csum;
}

// ---------------- gather inner tokens -> bf16 K-pair-packed A operand ----------------
__global__ void __launch_bounds__(128) k_gather_inner(
    const float* __restrict__ xin, const int* __restrict__ slot_src_i,
    unsigned int* __restrict__ A1) {
  int r = blockIdx.x;
  int src = slot_src_i[r];
  unsigned int* dst = A1 + (size_t)r * (D_ / 2);
  if (src >= 0) {
    const float* s = xin + (size_t)src * D_;
    for (int k = threadIdx.x; k < D_ / 2; k += 128)
      dst[k] = pack2bf(s[2 * k], s[2 * k + 1]);
  } else {
    for (int k = threadIdx.x; k < D_ / 2; k += 128) dst[k] = 0u;
  }
}

// ---------------- pack fp32 weights [E][K][N] -> bf16 pairs [E][K/2][N] ----------------
__global__ void __launch_bounds__(256) k_pack_w(
    const float* __restrict__ w, unsigned int* __restrict__ out, int K, int Nn) {
  size_t total  = (size_t)(EO_ * EI_) * (size_t)(K / 2) * (size_t)Nn;
  size_t stride = (size_t)gridDim.x * 256;
  for (size_t i = (size_t)blockIdx.x * 256 + threadIdx.x; i < total; i += stride) {
    size_t n  = i % (size_t)Nn;
    size_t k2 = (i / (size_t)Nn) % (size_t)(K / 2);
    size_t e  = i / ((size_t)Nn * (size_t)(K / 2));
    const float* base = w + (e * (size_t)K + 2 * k2) * (size_t)Nn + n;
    out[i] = pack2bf(base[0], base[Nn]);
  }
}

// ---------------- bf16 WMMA GEMM: C[E][M][N] = A[E][M][K] * B[E][K][N] ----------------
// A: K-pair packed dwords [E][M][K/2] staged by GLOBAL_LOAD_ASYNC_TO_LDS_B128.
// B: K-pair packed dwords [E][K/2][N] staged by the Tensor Data Mover (one D# per
//    block, LDS row padding 128dw+4dw done by the TDM itself).
// 256 threads = 8 waves; block tile 128x128; wave tile 32x64 (2x4 of 16x16).
template <bool RELU, bool OUT_BF16>
__global__ void __launch_bounds__(256) gemm_bf16_kernel(
    const unsigned int* __restrict__ A, const unsigned int* __restrict__ Bp,
    void* __restrict__ Out, int M, int K, int N) {
  const int e  = blockIdx.z;
  const int mb = blockIdx.y * 128;
  const int nb = blockIdx.x * 128;
  const int K2 = K >> 1;
  const unsigned int* Ae = A  + (size_t)e * (size_t)M  * K2;
  const unsigned int* Be = Bp + (size_t)e * (size_t)K2 * N;

  __shared__ unsigned int a_lds[128][20];   // [m][k2], 20-dword rows (16B aligned)
  __shared__ unsigned int b_lds[16][132];   // [k2][n], 132-dword rows (TDM padding)

  const int tid  = threadIdx.x;
  const int lane = tid & 31;
  const int wid  = tid >> 5;
  const int wm   = wid & 3;      // 4 wave rows  * 32
  const int wn   = wid >> 2;     // 2 wave cols  * 64
  const int lh   = lane & 15;
  const int hh   = lane >> 4;

  const unsigned a_base = (unsigned)(unsigned long long)&a_lds[0][0];
  const unsigned b_base = (unsigned)(unsigned long long)&b_lds[0][0];

  // A tile = 512 x 16B chunks; thread owns chunks {tid, tid+256}; OOB rows clamped
  // (duplicate row M-1 -> results for rows >= M are discarded in the epilogue).
  const int m0 = tid >> 2,         q0 = tid & 3;
  const int m1 = (tid + 256) >> 2, q1 = (tid + 256) & 3;
  const int row0 = (mb + m0 < M) ? (mb + m0) : (M - 1);
  const int row1 = (mb + m1 < M) ? (mb + m1) : (M - 1);
  const unsigned lds_a0 = a_base + (unsigned)(m0 * 20 + q0 * 4) * 4u;
  const unsigned lds_a1 = a_base + (unsigned)(m1 * 20 + q1 * 4) * 4u;

  v8f acc[2][4] = {};

  for (int kb2 = 0; kb2 < K2; kb2 += 16) {   // K-tile of 32 bf16
    // async DMA: A tile (per-lane B128), B tile (one TDM descriptor, wave 0)
    async_ld_b128(lds_a0, Ae + (size_t)row0 * K2 + kb2 + q0 * 4);
    async_ld_b128(lds_a1, Ae + (size_t)row1 * K2 + kb2 + q1 * 4);
    if (wid == 0) {
      tdm_load_2d(b_base, Be + (size_t)kb2 * N + nb,
                  /*tile_d0=*/128u, /*tile_d1=*/16u, /*stride0=*/(unsigned)N);
      __builtin_amdgcn_s_wait_tensorcnt(0);
    }
    if (kb2 + 16 < K2)  // pull next A tile toward the WGP (global_prefetch_b8)
      __builtin_prefetch(Ae + (size_t)row0 * K2 + kb2 + 16, 0, 0);
    wait_async0();
    __syncthreads();

    // A fragment: lanes 0-15 M=lane K=0..15; lanes 16-31 M=lane-16 K=16..31
    Frag16 af[2], bfr[4];
    #pragma unroll
    for (int t = 0; t < 2; ++t) {
      int m = wm * 32 + t * 16 + lh;
      #pragma unroll
      for (int j = 0; j < 8; ++j)
        af[t].u[j] = a_lds[m][j + hh * 8];
    }
    // B fragment: lanes 0-15 N=lane K=0..15; lanes 16-31 K=16..31
    #pragma unroll
    for (int t = 0; t < 4; ++t) {
      int n = wn * 64 + t * 16 + lh;
      #pragma unroll
      for (int j = 0; j < 8; ++j)
        bfr[t].u[j] = b_lds[j + hh * 8][n];
    }
    #pragma unroll
    for (int i = 0; i < 2; ++i)
      #pragma unroll
      for (int j = 0; j < 4; ++j)
        acc[i][j] = __builtin_amdgcn_wmma_f32_16x16x32_bf16(
            false, af[i].v, false, bfr[j].v, (short)0, acc[i][j], false, false);
    __syncthreads();
  }

  // epilogue: D layout — VGPR r holds (M=r, lanes 0-15) / (M=r+8, lanes 16-31), N=lane&15
  #pragma unroll
  for (int i = 0; i < 2; ++i) {
    #pragma unroll
    for (int j = 0; j < 4; ++j) {
      int gm0 = mb + wm * 32 + i * 16;
      int gn  = nb + wn * 64 + j * 16 + lh;
      #pragma unroll
      for (int r = 0; r < 8; ++r) {
        int row = gm0 + r + hh * 8;
        if (row < M) {
          float v = acc[i][j][r];
          if (RELU) v = v > 0.f ? v : 0.f;
          size_t off = (size_t)e * M * N + (size_t)row * N + gn;
          if (OUT_BF16) ((unsigned short*)Out)[off] = f2bf(v);
          else          ((float*)Out)[off] = v;
        }
      }
    }
  }
}

// ---------------- combine inner: out_o[slot] = g1*expout[r1] + g2*expout[r2] ----------------
__global__ void __launch_bounds__(256) k_combine_inner(
    const float* __restrict__ exp_out,
    const int* __restrict__ ir1, const int* __restrict__ ir2,
    const float* __restrict__ ig1, const float* __restrict__ ig2,
    float* __restrict__ out_o) {
  int row = blockIdx.x;
  int d = threadIdx.x * 4;
  int r1 = ir1[row], r2 = ir2[row];
  float a = ig1[row], c = ig2[row];
  float ax = 0.f, ay = 0.f, az = 0.f, aw = 0.f;
  if (r1 >= 0) {
    float4 v = *(const float4*)(exp_out + (size_t)r1 * D_ + d);
    ax += a * v.x; ay += a * v.y; az += a * v.z; aw += a * v.w;
  }
  if (r2 >= 0) {
    float4 v = *(const float4*)(exp_out + (size_t)r2 * D_ + d);
    ax += c * v.x; ay += c * v.y; az += c * v.z; aw += c * v.w;
  }
  *(float4*)(out_o + (size_t)row * D_ + d) = make_float4(ax, ay, az, aw);
}

// ---------------- combine outer: output token = g1*out_o[slot1] + g2*out_o[slot2] ----------------
__global__ void __launch_bounds__(256) k_combine_outer(
    const float* __restrict__ out_o,
    const int* __restrict__ e1, const int* __restrict__ e2,
    const int* __restrict__ p1, const int* __restrict__ p2,
    const float* __restrict__ g1, const float* __restrict__ g2,
    float* __restrict__ out) {
  int t = blockIdx.x;
  int b = t / N_;
  int d = threadIdx.x * 4;
  float ax = 0.f, ay = 0.f, az = 0.f, aw = 0.f;
  int q1 = p1[t];
  if (q1 >= 0) {
    float g = g1[t];
    float4 v = *(const float4*)(out_o + ((size_t)(e1[t] * B_ + b) * CAPO + q1) * D_ + d);
    ax += g * v.x; ay += g * v.y; az += g * v.z; aw += g * v.w;
  }
  int q2 = p2[t];
  if (q2 >= 0) {
    float g = g2[t];
    float4 v = *(const float4*)(out_o + ((size_t)(e2[t] * B_ + b) * CAPO + q2) * D_ + d);
    ax += g * v.x; ay += g * v.y; az += g * v.z; aw += g * v.w;
  }
  *(float4*)(out + (size_t)t * D_ + d) = make_float4(ax, ay, az, aw);
}

// ---------------- aux loss ----------------
__global__ void k_loss(const float* __restrict__ proxy_o, const float* __restrict__ cnt_o,
                       const float* __restrict__ proxy_i, const float* __restrict__ cnt_i,
                       float* __restrict__ out_loss) {
  if (threadIdx.x || blockIdx.x) return;
  float lo = 0.f;
  for (int i = 0; i < B_ * EO_; ++i) lo += (proxy_o[i] / N_) * (cnt_o[i] / N_);
  lo = lo / (float)(B_ * EO_) * (float)(EO_ * EO_);
  float li = 0.f;
  for (int i = 0; i < EO_ * B_ * EI_; ++i) li += (proxy_i[i] / CAPO) * (cnt_i[i] / CAPO);
  li = li / (float)(EO_ * B_ * EI_) * (float)(EI_ * EI_);
  out_loss[0] = (lo + li) * 0.01f;
}

// ---------------- host orchestration ----------------
extern "C" void kernel_launch(void* const* d_in, const int* in_sizes, int n_in,
                              void* d_out, int out_size, void* d_ws, size_t ws_size,
                              hipStream_t stream) {
  const float* x   = (const float*)d_in[0];  // [B,N,D]
  const float* wgo = (const float*)d_in[1];  // [D,EO]
  const float* wgi = (const float*)d_in[2];  // [EO,D,EI]
  const float* w1  = (const float*)d_in[3];  // [EO,EI,D,H]
  const float* w2  = (const float*)d_in[4];  // [EO,EI,H,D]
  float* out = (float*)d_out;                // [B,N,D] ++ [1] loss

  char* ws = (char*)d_ws;
  size_t off = 0;
  auto alloc = [&](size_t bytes) -> void* {
    void* p = ws + off;
    off += (bytes + 255) & ~(size_t)255;
    return p;
  };

  int*    tok_e1 = (int*)   alloc((size_t)TOK * 4);
  int*    tok_e2 = (int*)   alloc((size_t)TOK * 4);
  int*    tok_p1 = (int*)   alloc((size_t)TOK * 4);
  int*    tok_p2 = (int*)   alloc((size_t)TOK * 4);
  float*  tok_g1 = (float*) alloc((size_t)TOK * 4);
  float*  tok_g2 = (float*) alloc((size_t)TOK * 4);
  float4* tok_pr = (float4*)alloc((size_t)TOK * 16);
  float*  proxy_o = (float*)alloc((size_t)B_ * EO_ * 4);
  float*  cnt_o   = (float*)alloc((size_t)B_ * EO_ * 4);
  int*    slot_src_o  = (int*)  alloc((size_t)SLO * 4);
  float*  slot_gate_o = (float*)alloc((size_t)SLO * 4);
  int*    ie1 = (int*)   alloc((size_t)SLO * 4);
  int*    ie2 = (int*)   alloc((size_t)SLO * 4);
  float*  ig1 = (float*) alloc((size_t)SLO * 4);
  float*  ig2 = (float*) alloc((size_t)SLO * 4);
  float4* praw_i = (float4*)alloc((size_t)SLO * 16);
  int*    ir1 = (int*)   alloc((size_t)SLO * 4);
  int*    ir2 = (int*)   alloc((size_t)SLO * 4);
  int*    slot_src_i = (int*)alloc((size_t)SLI * 4);
  float*  proxy_i = (float*)alloc((size_t)EO_ * B_ * EI_ * 4);
  float*  cnt_i   = (float*)alloc((size_t)EO_ * B_ * EI_ * 4);
  float*  exp_in_o = (float*)alloc((size_t)SLO * D_ * 4);                    // 80 MiB
  unsigned int* A1  = (unsigned int*)alloc((size_t)SLI * (D_ / 2) * 4);      // 50 MiB
  unsigned int* w1p = (unsigned int*)alloc((size_t)16 * (D_ / 2) * H_ * 4);  // 128 MiB
  unsigned int* w2p = (unsigned int*)alloc((size_t)16 * (H_ / 2) * D_ * 4);  // 128 MiB
  unsigned short* hidden = (unsigned short*)alloc((size_t)16 * MR * H_ * 2); // 200 MiB
  float* exp_out = (float*)alloc((size_t)16 * MR * D_ * 4);                  // 100 MiB
  float* out_o   = (float*)alloc((size_t)SLO * D_ * 4);                      // 80 MiB
  (void)in_sizes; (void)n_in; (void)out_size; (void)ws_size;

  k_init<<<(SLI + 255) / 256, 256, 0, stream>>>(slot_src_o, slot_gate_o, ir1, ir2, slot_src_i);

  k_outer_gate<<<(TOK + 127) / 128, 128, 0, stream>>>(
      x, wgo, tok_e1, tok_e2, tok_g1, tok_g2, tok_pr);
  k_outer_scan<<<B_, 32, 0, stream>>>(
      tok_e1, tok_e2, tok_g1, tok_g2, tok_pr,
      tok_p1, tok_p2, slot_src_o, slot_gate_o, proxy_o, cnt_o);
  k_gather_outer<<<SLO, 256, 0, stream>>>(x, slot_src_o, exp_in_o);

  k_inner_gate<<<(SLO + 127) / 128, 128, 0, stream>>>(
      exp_in_o, wgi, slot_gate_o, ie1, ie2, ig1, ig2, praw_i);
  k_inner_scan<<<EO_ * B_, 32, 0, stream>>>(
      ie1, ie2, ig2, praw_i, ir1, ir2, slot_src_i, proxy_i, cnt_i);
  k_gather_inner<<<SLI, 128, 0, stream>>>(exp_in_o, slot_src_i, A1);

  k_pack_w<<<8192, 256, 0, stream>>>(w1, w1p, D_, H_);
  k_pack_w<<<8192, 256, 0, stream>>>(w2, w2p, H_, D_);

  // FFN: hidden = relu(A1 * w1)  [bf16 out],  exp_out = hidden * w2  [f32 out]
  gemm_bf16_kernel<true, true><<<dim3(H_ / 128, (MR + 127) / 128, 16), 256, 0, stream>>>(
      A1, w1p, (void*)hidden, MR, D_, H_);
  gemm_bf16_kernel<false, false><<<dim3(D_ / 128, (MR + 127) / 128, 16), 256, 0, stream>>>(
      (const unsigned int*)hidden, w2p, (void*)exp_out, MR, H_, D_);

  k_combine_inner<<<SLO, 256, 0, stream>>>(exp_out, ir1, ir2, ig1, ig2, out_o);
  k_combine_outer<<<TOK, 256, 0, stream>>>(
      out_o, tok_e1, tok_e2, tok_p1, tok_p2, tok_g1, tok_g2, out);
  k_loss<<<1, 32, 0, stream>>>(proxy_o, cnt_o, proxy_i, cnt_i, out + (size_t)TOK * D_);
}
